// PointTransformerCls_27547920236581
// MI455X (gfx1250) — matrix operand order repaired and checked by reading, not compiled, not run on hardware
//
#include <hip/hip_runtime.h>
#include <hip/hip_bf16.h>
#include <math.h>

// ---------------------------------------------------------------------------
// CDNA5 (gfx1250) wave32 WMMA types
// ---------------------------------------------------------------------------
typedef __attribute__((ext_vector_type(16))) __bf16 v16bf;
typedef __attribute__((ext_vector_type(8)))  float  v8f;

#define BN_EPS 1e-5f

// ---------------------------------------------------------------------------
// GEMM: out[M,N] = epi( A[M,K] @ W[K,N] ),  epi = (+bias)->(BN)->(+res)->(ReLU)
// REQUIRES: M % 64 == 0, N % 32 == 0, K % 32 == 0 (true for every call site).
// Block = 128 threads = 4 waves. Block tile = 64(M) x 32(N).
// Each wave owns a 16x32 output strip: per K-step it loads ONE A fragment
// (4x b128, contiguous) and issues TWO v_wmma_f32_16x16x32_bf16 against the
// 32x32 B tile staged in LDS (shared by all 4 waves -> 4x B reuse, and the A
// fragment is reused across both WMMAs).
// ---------------------------------------------------------------------------
__global__ __launch_bounds__(128) void gemm_wmma_bf16(
    const float* __restrict__ A, const float* __restrict__ W,
    const float* __restrict__ bias,
    const float* __restrict__ bn_g, const float* __restrict__ bn_b,
    const float* __restrict__ bn_m, const float* __restrict__ bn_v,
    const float* __restrict__ res, float* __restrict__ out,
    int M, int N, int K, int relu)
{
  __shared__ __align__(16) float bs[32 * 32];   // B tile, [k][n], 4KB

  const int tid  = threadIdx.x;
  const int wave = tid >> 5;
  const int lane = tid & 31;
  const int col0 = blockIdx.x * 32;
  const int row0 = (blockIdx.y * 4 + wave) * 16;
  const int half = lane >> 4;                   // 0: lanes 0-15, 1: lanes 16-31
  const int l15  = lane & 15;
  const int arow = row0 + l15;

  // B staging: 32 rows x 32 cols = 1024 floats; 128 threads x 2 float4 each.
  const int br  = tid >> 2;                     // 0..31 (k within tile)
  const int bc8 = (tid & 3) * 8;                // 0,8,16,24 (n within tile)

  v8f acc0 = {};
  v8f acc1 = {};
  for (int k0 = 0; k0 < K; k0 += 32) {
    // ---- stage B tile (coalesced rows, 64B aligned) ----
    {
      const float* wrow = &W[(size_t)(k0 + br) * N + col0 + bc8];
      const float4 w0 = *(const float4*)(wrow + 0);
      const float4 w1 = *(const float4*)(wrow + 4);
      *(float4*)&bs[br * 32 + bc8 + 0] = w0;
      *(float4*)&bs[br * 32 + bc8 + 4] = w1;
    }
    __syncthreads();

    // ---- A fragment: two contiguous 8-float runs per lane ----
    const float* arp = &A[(size_t)arow * K + k0 + half * 8];
    const float4 a0 = *(const float4*)(arp + 0);
    const float4 a1 = *(const float4*)(arp + 4);
    const float4 a2 = *(const float4*)(arp + 16);
    const float4 a3 = *(const float4*)(arp + 20);
    if (k0 + 32 < K) __builtin_prefetch(arp + 32, 0, 0);

    v16bf a;
    a[0]  = (__bf16)a0.x; a[1]  = (__bf16)a0.y; a[2]  = (__bf16)a0.z; a[3]  = (__bf16)a0.w;
    a[4]  = (__bf16)a1.x; a[5]  = (__bf16)a1.y; a[6]  = (__bf16)a1.z; a[7]  = (__bf16)a1.w;
    a[8]  = (__bf16)a2.x; a[9]  = (__bf16)a2.y; a[10] = (__bf16)a2.z; a[11] = (__bf16)a2.w;
    a[12] = (__bf16)a3.x; a[13] = (__bf16)a3.y; a[14] = (__bf16)a3.z; a[15] = (__bf16)a3.w;

    // ---- B fragments from LDS: rows = K (half-selected), col = l15 (+16) ----
    v16bf b0, b1;
    #pragma unroll
    for (int v = 0; v < 16; ++v) {
      const int krow = ((v < 8) ? (half * 8 + v) : (16 + half * 8 + (v - 8))) * 32;
      b0[v] = (__bf16)bs[krow + l15];
      b1[v] = (__bf16)bs[krow + 16 + l15];
    }

    acc0 = __builtin_amdgcn_wmma_f32_16x16x32_bf16(
        false, a, false, b0, (short)0, acc0, false, false);
    acc1 = __builtin_amdgcn_wmma_f32_16x16x32_bf16(
        false, a, false, b1, (short)0, acc1, false, false);
    __syncthreads();
  }

  // ---- epilogue; C/D layout: M = r + 8*half, N = l15 (exact tiles) ----
  #pragma unroll
  for (int t = 0; t < 2; ++t) {
    const int ni = col0 + t * 16 + l15;
    float fb = bias ? bias[ni] : 0.f;
    float scale = 1.f, shift = 0.f;
    if (bn_g) {
      scale = rsqrtf(bn_v[ni] + BN_EPS) * bn_g[ni];
      shift = bn_b[ni] - bn_m[ni] * scale;
    }
    #pragma unroll
    for (int r = 0; r < 8; ++r) {
      const int mi = row0 + r + half * 8;
      float v = (t == 0 ? acc0[r] : acc1[r]) + fb;
      if (bn_g) v = v * scale + shift;
      if (res)  v += res[(size_t)mi * N + ni];
      if (relu) v = fmaxf(v, 0.f);
      out[(size_t)mi * N + ni] = v;
    }
  }
}

// ---------------------------------------------------------------------------
// concat xyz + feat -> x6 [B*n0, 6]
// ---------------------------------------------------------------------------
__global__ void concat6_kernel(const float* __restrict__ p, const float* __restrict__ x,
                               float* __restrict__ x6, int rows)
{
  int e = blockIdx.x * blockDim.x + threadIdx.x;
  if (e >= rows * 6) return;
  int i = e / 6, c = e % 6;
  x6[e] = (c < 3) ? p[i * 3 + c] : x[i * 3 + (c - 3)];
}

// ---------------------------------------------------------------------------
// stride-1 transition down: relu(bn(x6 @ W))   (K=6, tiny -> scalar)
// ---------------------------------------------------------------------------
__global__ void td1_kernel(const float* __restrict__ x6, const float* __restrict__ W,
                           const float* __restrict__ g, const float* __restrict__ b,
                           const float* __restrict__ m, const float* __restrict__ v,
                           float* __restrict__ out, int rows, int C)
{
  int e = blockIdx.x * blockDim.x + threadIdx.x;
  if (e >= rows * C) return;
  int i = e / C, c = e % C;
  float s = 0.f;
  #pragma unroll
  for (int t = 0; t < 6; ++t) s += x6[i * 6 + t] * W[t * C + c];
  s = (s - m[c]) * rsqrtf(v[c] + BN_EPS) * g[c] + b[c];
  out[e] = fmaxf(s, 0.f);
}

// ---------------------------------------------------------------------------
// Furthest point sampling: one block per batch, dist array in LDS.
// ---------------------------------------------------------------------------
__global__ __launch_bounds__(256) void fps_kernel(const float* __restrict__ p,
                                                  int n, int m, int* __restrict__ sel)
{
  extern __shared__ char smem[];
  float* dist = (float*)smem;              // n floats
  __shared__ float rv[256];
  __shared__ int   ra[256];
  __shared__ int   s_cur;
  const int b = blockIdx.x;
  const float* pb = p + (size_t)b * n * 3;
  const int tid = threadIdx.x;

  for (int j = tid; j < n; j += blockDim.x) dist[j] = 3.4e38f;
  if (tid == 0) s_cur = 0;
  __syncthreads();

  for (int i = 0; i < m; ++i) {
    const int cur = s_cur;
    if (tid == 0) sel[(size_t)b * m + i] = cur;
    const float cx = pb[cur * 3 + 0], cy = pb[cur * 3 + 1], cz = pb[cur * 3 + 2];
    float bestv = -1.f; int besta = 0;
    for (int j = tid; j < n; j += blockDim.x) {
      float dx = pb[j * 3 + 0] - cx, dy = pb[j * 3 + 1] - cy, dz = pb[j * 3 + 2] - cz;
      float d = fminf(dist[j], dx * dx + dy * dy + dz * dz);
      dist[j] = d;
      if (d > bestv) { bestv = d; besta = j; }
    }
    rv[tid] = bestv; ra[tid] = besta;
    __syncthreads();
    for (int s = 128; s > 0; s >>= 1) {
      if (tid < s && rv[tid + s] > rv[tid]) { rv[tid] = rv[tid + s]; ra[tid] = ra[tid + s]; }
      __syncthreads();
    }
    if (tid == 0) s_cur = ra[0];
    __syncthreads();
  }
}

// ---------------------------------------------------------------------------
// gather sampled coordinates: np[b,i,:] = p[b, sel[b,i], :]
// ---------------------------------------------------------------------------
__global__ void gather_np_kernel(const float* __restrict__ p, const int* __restrict__ sel,
                                 float* __restrict__ np_, int n, int m, int total3)
{
  int e = blockIdx.x * blockDim.x + threadIdx.x;
  if (e >= total3) return;
  int g = e / 3, c = e % 3;
  int b = g / m;
  np_[e] = p[((size_t)b * n + sel[g]) * 3 + c];
}

// ---------------------------------------------------------------------------
// Brute-force kNN: one thread per query, register top-k insertion sort.
// ---------------------------------------------------------------------------
template <int KNN>
__global__ __launch_bounds__(256) void knn_kernel(const float* __restrict__ q,
                                                  const float* __restrict__ x,
                                                  int m, int n, int Bm,
                                                  int* __restrict__ out)
{
  int g = blockIdx.x * blockDim.x + threadIdx.x;
  if (g >= Bm) return;
  int b = g / m;
  const float* qp = q + (size_t)g * 3;
  const float* xb = x + (size_t)b * n * 3;
  const float qx = qp[0], qy = qp[1], qz = qp[2];
  float bd[KNN]; int bi[KNN];
  #pragma unroll
  for (int t = 0; t < KNN; ++t) { bd[t] = 3.4e38f; bi[t] = 0; }
  for (int j = 0; j < n; ++j) {
    float dx = xb[j * 3 + 0] - qx, dy = xb[j * 3 + 1] - qy, dz = xb[j * 3 + 2] - qz;
    float d = dx * dx + dy * dy + dz * dz;
    if (d < bd[KNN - 1]) {
      int t = KNN - 1;
      while (t > 0 && bd[t - 1] > d) { bd[t] = bd[t - 1]; bi[t] = bi[t - 1]; --t; }
      bd[t] = d; bi[t] = j;
    }
  }
  #pragma unroll
  for (int t = 0; t < KNN; ++t) out[(size_t)g * KNN + t] = bi[t];
}

// ---------------------------------------------------------------------------
// stride-4 transition down: gather neighbor feats into LDS, then
// per output channel: max over ns of relu(bn(feat @ W)).
// Block = one output point, blockDim.x = C.
// ---------------------------------------------------------------------------
struct TD4Args {
  const float* p; const float* np_; const float* x; const int* idx;
  const float* W;
  const float* bn_g; const float* bn_b; const float* bn_m; const float* bn_v;
  float* out;
  int n, m, ns, Cin, C;
};

__global__ void td4_kernel(TD4Args a)
{
  extern __shared__ float feat[];            // ns * (3 + Cin)
  const int gi = blockIdx.x;                 // 0..B*m-1
  const int b  = gi / a.m;
  const int Kf = 3 + a.Cin;
  const int tid = threadIdx.x;

  for (int e = tid; e < a.ns * Kf; e += blockDim.x) {
    int j = e / Kf, c = e % Kf;
    int gj = b * a.n + a.idx[(size_t)gi * a.ns + j];
    feat[e] = (c < 3) ? (a.p[(size_t)gj * 3 + c] - a.np_[(size_t)gi * 3 + c])
                      : a.x[(size_t)gj * a.Cin + (c - 3)];
  }
  __syncthreads();

  const int co = tid;
  if (co < a.C) {
    const float sc = rsqrtf(a.bn_v[co] + BN_EPS) * a.bn_g[co];
    float mx = -3.4e38f;
    for (int j = 0; j < a.ns; ++j) {
      float s = 0.f;
      for (int k = 0; k < Kf; ++k) s += feat[j * Kf + k] * a.W[(size_t)k * a.C + co];
      s = (s - a.bn_m[co]) * sc + a.bn_b[co];
      mx = fmaxf(mx, fmaxf(s, 0.f));
    }
    a.out[(size_t)gi * a.C + co] = mx;
  }
}

// ---------------------------------------------------------------------------
// Fused point-transformer attention (everything after q/k/v GEMMs).
// One block per point; all intermediates live in LDS. Includes bn2+ReLU.
// ---------------------------------------------------------------------------
struct AttnArgs {
  const float* p; const int* idx;
  const float* xq; const float* xk; const float* xv;
  const float* p0w; const float* p0b;
  const float* pbn_g; const float* pbn_b; const float* pbn_m; const float* pbn_v;
  const float* p3w; const float* p3b;
  const float* bn0_g; const float* bn0_b; const float* bn0_m; const float* bn0_v;
  const float* w2w; const float* w2b;
  const float* bn3_g; const float* bn3_b; const float* bn3_m; const float* bn3_v;
  const float* w5w; const float* w5b;
  const float* bn2_g; const float* bn2_b; const float* bn2_m; const float* bn2_v;
  float* out;
  int n, ns, C, Cs;
};

__global__ __launch_bounds__(256) void attn_kernel(AttnArgs a)
{
  extern __shared__ char smem[];
  const int ns = a.ns, C = a.C, Cs = a.Cs;
  float* pr   = (float*)smem;            // ns*C   relative-position embedding
  float* wa   = pr + ns * C;             // ns*C   bn0/relu'd logits input
  float* hh   = wa + ns * C;             // ns*Cs  w2 hidden
  float* sm   = hh + ns * Cs;            // ns*Cs  logits -> softmax
  float* xqv  = sm + ns * Cs;            // C
  float* pr3  = xqv + C;                 // ns*3
  int*   idxs = (int*)(pr3 + ns * 3);    // ns

  const int gi = blockIdx.x;
  const int b  = gi / a.n;
  const int tid = threadIdx.x;

  if (tid < ns) idxs[tid] = a.idx[(size_t)gi * ns + tid];
  for (int c = tid; c < C; c += blockDim.x) xqv[c] = a.xq[(size_t)gi * C + c];
  __syncthreads();

  // p_r (3d): (p_j - p_i) @ p0w + p0b
  for (int e = tid; e < ns * 3; e += blockDim.x) {
    int j = e / 3, c = e % 3;
    int gj = b * a.n + idxs[j];
    float s = a.p0b[c];
    #pragma unroll
    for (int t = 0; t < 3; ++t)
      s += (a.p[(size_t)gj * 3 + t] - a.p[(size_t)gi * 3 + t]) * a.p0w[t * 3 + c];
    pr3[e] = s;
  }
  __syncthreads();

  // pr = relu(bn(pr3)) @ p3w + p3b     [ns, C]
  for (int e = tid; e < ns * C; e += blockDim.x) {
    int j = e / C, c = e % C;
    float s = a.p3b[c];
    #pragma unroll
    for (int t = 0; t < 3; ++t) {
      float u = pr3[j * 3 + t];
      u = (u - a.pbn_m[t]) * rsqrtf(a.pbn_v[t] + BN_EPS) * a.pbn_g[t] + a.pbn_b[t];
      s += fmaxf(u, 0.f) * a.p3w[t * C + c];
    }
    pr[e] = s;
  }
  __syncthreads();

  // wa = relu(bn0(xk_j - xq_i + pr))
  for (int e = tid; e < ns * C; e += blockDim.x) {
    int j = e / C, c = e % C;
    int gj = b * a.n + idxs[j];
    float u = a.xk[(size_t)gj * C + c] - xqv[c] + pr[e];
    u = (u - a.bn0_m[c]) * rsqrtf(a.bn0_v[c] + BN_EPS) * a.bn0_g[c] + a.bn0_b[c];
    wa[e] = fmaxf(u, 0.f);
  }
  __syncthreads();

  // hh = relu(bn3(wa @ w2 + w2b))      [ns, Cs]
  for (int e = tid; e < ns * Cs; e += blockDim.x) {
    int j = e / Cs, c2 = e % Cs;
    float s = a.w2b[c2];
    for (int c = 0; c < C; ++c) s += wa[j * C + c] * a.w2w[(size_t)c * Cs + c2];
    s = (s - a.bn3_m[c2]) * rsqrtf(a.bn3_v[c2] + BN_EPS) * a.bn3_g[c2] + a.bn3_b[c2];
    hh[e] = fmaxf(s, 0.f);
  }
  __syncthreads();

  // logits = hh @ w5 + w5b             [ns, Cs]
  for (int e = tid; e < ns * Cs; e += blockDim.x) {
    int j = e / Cs, c2 = e % Cs;
    float s = a.w5b[c2];
    for (int t = 0; t < Cs; ++t) s += hh[j * Cs + t] * a.w5w[t * Cs + c2];
    sm[e] = s;
  }
  __syncthreads();

  // softmax over neighbors (axis j), per channel c2
  for (int c2 = tid; c2 < Cs; c2 += blockDim.x) {
    float mx = -3.4e38f;
    for (int j = 0; j < ns; ++j) mx = fmaxf(mx, sm[j * Cs + c2]);
    float ssum = 0.f;
    for (int j = 0; j < ns; ++j) {
      float ev = __expf(sm[j * Cs + c2] - mx);
      sm[j * Cs + c2] = ev; ssum += ev;
    }
    float inv = 1.f / ssum;
    for (int j = 0; j < ns; ++j) sm[j * Cs + c2] *= inv;
  }
  __syncthreads();

  // out[c] = relu(bn2( sum_j (xv_j[c] + pr[j,c]) * sm[j, c % Cs] ))
  for (int c = tid; c < C; c += blockDim.x) {
    const int c2 = c % Cs;
    float s = 0.f;
    for (int j = 0; j < ns; ++j) {
      int gj = b * a.n + idxs[j];
      s += (a.xv[(size_t)gj * C + c] + pr[j * C + c]) * sm[j * Cs + c2];
    }
    s = (s - a.bn2_m[c]) * rsqrtf(a.bn2_v[c] + BN_EPS) * a.bn2_g[c] + a.bn2_b[c];
    a.out[(size_t)gi * C + c] = fmaxf(s, 0.f);
  }
}

// ---------------------------------------------------------------------------
// mean pool over points per segment
// ---------------------------------------------------------------------------
__global__ void meanpool_kernel(const float* __restrict__ x, float* __restrict__ out,
                                int B, int n, int C)
{
  int e = blockIdx.x * blockDim.x + threadIdx.x;
  if (e >= B * C) return;
  int b = e / C, c = e % C;
  float s = 0.f;
  for (int i = 0; i < n; ++i) s += x[((size_t)b * n + i) * C + c];
  out[e] = s / (float)n;
}

// ---------------------------------------------------------------------------
// head: relu(bn(x@W1+b1)) @ W2 + b2   (512->256->40), one block per batch row
// ---------------------------------------------------------------------------
struct HeadArgs {
  const float* pooled;
  const float* w1; const float* b1;
  const float* bn_g; const float* bn_b; const float* bn_m; const float* bn_v;
  const float* w2; const float* b2;
  float* out;
};

__global__ __launch_bounds__(256) void head_kernel(HeadArgs a)
{
  __shared__ float h[256];
  const int b = blockIdx.x, tid = threadIdx.x;
  const float* xin = a.pooled + (size_t)b * 512;
  float s = a.b1[tid];
  for (int k = 0; k < 512; ++k) s += xin[k] * a.w1[k * 256 + tid];
  s = (s - a.bn_m[tid]) * rsqrtf(a.bn_v[tid] + BN_EPS) * a.bn_g[tid] + a.bn_b[tid];
  h[tid] = fmaxf(s, 0.f);
  __syncthreads();
  if (tid < 40) {
    float o = a.b2[tid];
    for (int k = 0; k < 256; ++k) o += h[k] * a.w2[k * 40 + tid];
    a.out[(size_t)b * 40 + tid] = o;
  }
}

// ---------------------------------------------------------------------------
// Host-side parameter views (pytree flattened in sorted-key order)
// ---------------------------------------------------------------------------
struct HBN  { const float *b, *g, *m, *v; };
struct HLin { const float *b, *w; };
struct HTr  { HLin k, p0, p3; HBN p_bn; HLin q, v, w2, w5; HBN w_bn0, w_bn3; };
struct HBlock { HBN bn1, bn2, bn3; HLin lin1, lin3; HTr tr; };
struct HEnc { HBlock blk; HBN td_bn; HLin td_lin; };

extern "C" void kernel_launch(void* const* d_in, const int* in_sizes, int n_in,
                              void* d_out, int out_size, void* d_ws, size_t ws_size,
                              hipStream_t stream)
{
  (void)in_sizes; (void)n_in; (void)out_size; (void)ws_size;
  const int B = 8, N0 = 4096;
  const int PLANES[5]  = {32, 64, 128, 256, 512};
  const int NSAMP[5]   = {8, 16, 16, 16, 16};

  const float* in_p = (const float*)d_in[0];
  const float* in_x = (const float*)d_in[1];
  // d_in[2] = o (segment offsets) unused: segments are equal-size

  // ---- consume the params pytree (sorted-key flatten order) ----
  int cur = 3;
  auto nf = [&]() -> const float* { return (const float*)d_in[cur++]; };
  auto rd_bn  = [&]() { HBN r; r.b = nf(); r.g = nf(); r.m = nf(); r.v = nf(); return r; };
  auto rd_lin = [&](bool bias) { HLin r; r.b = bias ? nf() : nullptr; r.w = nf(); return r; };

  HEnc enc[5];
  for (int i = 0; i < 5; ++i) {
    // enc dict keys sorted: "blocks" < "td"
    HBlock& bl = enc[i].blk;                // single block per stage
    bl.bn1 = rd_bn(); bl.bn2 = rd_bn(); bl.bn3 = rd_bn();
    bl.lin1 = rd_lin(false); bl.lin3 = rd_lin(false);
    HTr& t = bl.tr;                         // keys: k,p0,p3,p_bn,q,v,w2,w5,w_bn0,w_bn3
    t.k = rd_lin(true); t.p0 = rd_lin(true); t.p3 = rd_lin(true);
    t.p_bn = rd_bn();
    t.q = rd_lin(true); t.v = rd_lin(true); t.w2 = rd_lin(true); t.w5 = rd_lin(true);
    t.w_bn0 = rd_bn(); t.w_bn3 = rd_bn();
    enc[i].td_bn = rd_bn(); enc[i].td_lin = rd_lin(false);
  }
  HBN  head_bn = rd_bn();
  HLin head_l1 = rd_lin(true);
  HLin head_l2 = rd_lin(true);

  // ---- workspace bump allocator ----
  char* wp = (char*)d_ws;
  auto bumpf = [&](size_t cnt) { float* r = (float*)wp; wp += ((cnt * 4 + 255) & ~(size_t)255); return r; };
  auto bumpi = [&](size_t cnt) { int*   r = (int*)wp;   wp += ((cnt * 4 + 255) & ~(size_t)255); return r; };

  const size_t MAXRC = (size_t)B * N0 * 32;          // largest rows*C over all stages
  float* x6  = bumpf((size_t)B * N0 * 6);
  float* xF  = bumpf(MAXRC);
  float* xG  = bumpf(MAXRC);
  float* t1  = bumpf(MAXRC);
  float* t2  = bumpf(MAXRC);
  float* xq  = bumpf(MAXRC);
  float* xk  = bumpf(MAXRC);
  float* xv  = bumpf(MAXRC);
  float* pb0 = bumpf((size_t)B * 1024 * 3);
  float* pb1 = bumpf((size_t)B * 1024 * 3);
  int* sel    = bumpi((size_t)B * 1024);
  int* idxTD  = bumpi((size_t)B * 1024 * 16);
  int* idxA   = bumpi((size_t)B * N0 * 8);
  float* pooled = bumpf((size_t)B * 512);

  auto gemm = [&](const float* A, const float* W, const float* bias, const HBN* bn,
                  const float* res, float* out, int M, int N, int K, int relu) {
    dim3 grid(N / 32, M / 64);               // exact tiles (M%64==0, N%32==0)
    gemm_wmma_bf16<<<grid, 128, 0, stream>>>(
        A, W, bias,
        bn ? bn->g : nullptr, bn ? bn->b : nullptr,
        bn ? bn->m : nullptr, bn ? bn->v : nullptr,
        res, out, M, N, K, relu);
  };

  auto knn = [&](const float* q, const float* x, int m, int n, int k, int* out) {
    int total = B * m;
    int blocks = (total + 255) / 256;
    if (k == 8) knn_kernel<8><<<blocks, 256, 0, stream>>>(q, x, m, n, total, out);
    else        knn_kernel<16><<<blocks, 256, 0, stream>>>(q, x, m, n, total, out);
  };

  auto run_block = [&](const HBlock& bl, const float* p, int n, int C, int ns) {
    const int rows = B * n;
    const int Cs = C / 8;
    // lin1 + bn1 + relu
    gemm(xF, bl.lin1.w, nullptr, &bl.bn1, nullptr, t1, rows, C, C, 1);
    // q / k / v
    gemm(t1, bl.tr.q.w, bl.tr.q.b, nullptr, nullptr, xq, rows, C, C, 0);
    gemm(t1, bl.tr.k.w, bl.tr.k.b, nullptr, nullptr, xk, rows, C, C, 0);
    gemm(t1, bl.tr.v.w, bl.tr.v.b, nullptr, nullptr, xv, rows, C, C, 0);
    // fused attention (+ bn2 + relu) -> t2
    AttnArgs aa;
    aa.p = p; aa.idx = idxA; aa.xq = xq; aa.xk = xk; aa.xv = xv;
    aa.p0w = bl.tr.p0.w; aa.p0b = bl.tr.p0.b;
    aa.pbn_g = bl.tr.p_bn.g; aa.pbn_b = bl.tr.p_bn.b; aa.pbn_m = bl.tr.p_bn.m; aa.pbn_v = bl.tr.p_bn.v;
    aa.p3w = bl.tr.p3.w; aa.p3b = bl.tr.p3.b;
    aa.bn0_g = bl.tr.w_bn0.g; aa.bn0_b = bl.tr.w_bn0.b; aa.bn0_m = bl.tr.w_bn0.m; aa.bn0_v = bl.tr.w_bn0.v;
    aa.w2w = bl.tr.w2.w; aa.w2b = bl.tr.w2.b;
    aa.bn3_g = bl.tr.w_bn3.g; aa.bn3_b = bl.tr.w_bn3.b; aa.bn3_m = bl.tr.w_bn3.m; aa.bn3_v = bl.tr.w_bn3.v;
    aa.w5w = bl.tr.w5.w; aa.w5b = bl.tr.w5.b;
    aa.bn2_g = bl.bn2.g; aa.bn2_b = bl.bn2.b; aa.bn2_m = bl.bn2.m; aa.bn2_v = bl.bn2.v;
    aa.out = t2; aa.n = n; aa.ns = ns; aa.C = C; aa.Cs = Cs;
    size_t lds = (size_t)(2 * ns * C + 2 * ns * Cs + C + ns * 3) * sizeof(float)
               + (size_t)ns * sizeof(int);
    attn_kernel<<<rows, 256, lds, stream>>>(aa);
    // lin3 + bn3 + residual(xF) + relu -> xF (per-element read-then-write, safe)
    gemm(t2, bl.lin3.w, nullptr, &bl.bn3, xF, xF, rows, C, C, 1);
  };

  // ================= stage 1 (stride 1) =================
  const float* p_cur = in_p;
  int n_cur = N0;
  int Cin = 6;
  {
    const int C = PLANES[0], ns = NSAMP[0];
    const int rows = B * N0;
    concat6_kernel<<<(rows * 6 + 255) / 256, 256, 0, stream>>>(in_p, in_x, x6, rows);
    td1_kernel<<<(rows * C + 255) / 256, 256, 0, stream>>>(
        x6, enc[0].td_lin.w,
        enc[0].td_bn.g, enc[0].td_bn.b, enc[0].td_bn.m, enc[0].td_bn.v,
        xF, rows, C);
    knn(p_cur, p_cur, n_cur, n_cur, ns, idxA);
    run_block(enc[0].blk, p_cur, n_cur, C, ns);
    Cin = C;
  }

  // ================= stages 2..5 (stride 4) =================
  for (int i = 1; i < 5; ++i) {
    const int C = PLANES[i], ns = NSAMP[i];
    const int m = n_cur / 4;
    float* pnext = (i & 1) ? pb0 : pb1;

    // FPS + coordinate gather
    fps_kernel<<<B, 256, (size_t)n_cur * sizeof(float), stream>>>(p_cur, n_cur, m, sel);
    gather_np_kernel<<<(B * m * 3 + 255) / 256, 256, 0, stream>>>(
        p_cur, sel, pnext, n_cur, m, B * m * 3);

    // kNN of sampled points against previous cloud, then TD pool
    knn(pnext, p_cur, m, n_cur, ns, idxTD);
    TD4Args td;
    td.p = p_cur; td.np_ = pnext; td.x = xF; td.idx = idxTD;
    td.W = enc[i].td_lin.w;
    td.bn_g = enc[i].td_bn.g; td.bn_b = enc[i].td_bn.b;
    td.bn_m = enc[i].td_bn.m; td.bn_v = enc[i].td_bn.v;
    td.out = xG; td.n = n_cur; td.m = m; td.ns = ns; td.Cin = Cin; td.C = C;
    td4_kernel<<<B * m, C, (size_t)ns * (3 + Cin) * sizeof(float), stream>>>(td);

    { float* tmp = xF; xF = xG; xG = tmp; }
    p_cur = pnext; n_cur = m;

    // self-kNN for attention, then the transformer block
    knn(p_cur, p_cur, m, m, ns, idxA);
    run_block(enc[i].blk, p_cur, n_cur, C, ns);
    Cin = C;
  }

  // ================= mean pool + head =================
  meanpool_kernel<<<(B * 512 + 255) / 256, 256, 0, stream>>>(xF, pooled, B, n_cur, 512);
  HeadArgs ha;
  ha.pooled = pooled;
  ha.w1 = head_l1.w; ha.b1 = head_l1.b;
  ha.bn_g = head_bn.g; ha.bn_b = head_bn.b; ha.bn_m = head_bn.m; ha.bn_v = head_bn.v;
  ha.w2 = head_l2.w; ha.b2 = head_l2.b;
  ha.out = (float*)d_out;
  head_kernel<<<B, 256, 0, stream>>>(ha);
}
